// Mem_Conv2d_13683765805728
// MI455X (gfx1250) — matrix operand order repaired and checked by reading, not compiled, not run on hardware
//
#include <hip/hip_runtime.h>
#include <hip/hip_bf16.h>

typedef __attribute__((ext_vector_type(16))) _Float16 v16h;
typedef __attribute__((ext_vector_type(2)))  __fp16   h2;   // return type of cvt_pkrtz
typedef __attribute__((ext_vector_type(8)))  float    v8f;

#define NCH   32          // in/out channels
#define HW    256         // spatial
#define XW    258         // padded row length in LDS (1-col halo each side)
#define NW    9216        // weight element count: 32*32*3*3

// ---------------------------------------------------------------------------
// Kernel 1: wmax = max|W| + 1e-12  (single block reduction)
// ---------------------------------------------------------------------------
__global__ __launch_bounds__(256) void wmax_kernel(const float* __restrict__ w,
                                                   float* __restrict__ wmaxp) {
    __shared__ float red[256];
    float m = 0.0f;
    for (int i = threadIdx.x; i < NW; i += 256) m = fmaxf(m, fabsf(w[i]));
    red[threadIdx.x] = m;
    __syncthreads();
    for (int s = 128; s > 0; s >>= 1) {
        if (threadIdx.x < s) red[threadIdx.x] = fmaxf(red[threadIdx.x], red[threadIdx.x + s]);
        __syncthreads();
    }
    if (threadIdx.x == 0) wmaxp[0] = red[0] + 1e-12f;
}

// ---------------------------------------------------------------------------
// Kernel 2: quantize (4-bit symmetric, RNE like jnp.round) and pre-swizzle the
// weights into WMMA A-fragment order:
//   apack[((t*9 + j)*32 + lane)*16 + i] = f16(qW[co][ci][kh][kw])
// with co = t*16 + (lane&15), j = kh*3+kw,
//      ci = i + (lane>=16 ? 16 : 0)   (contiguous K order, matches B)
// ---------------------------------------------------------------------------
__global__ __launch_bounds__(256) void pack_kernel(const float* __restrict__ w,
                                                   const float* __restrict__ wmaxp,
                                                   _Float16* __restrict__ apack) {
    int tid = blockIdx.x * 256 + threadIdx.x;
    if (tid >= NW) return;
    int i  = tid & 15;
    int L  = (tid >> 4) & 31;
    int jt = tid >> 9;            // = t*9 + j   (32*16 = 512)
    int j  = jt % 9;
    int t  = jt / 9;
    int co = t * 16 + (L & 15);
    int ci = i + ((L >= 16) ? 16 : 0);
    int kh = j / 3, kw = j % 3;
    float wmax = wmaxp[0];
    float wv = w[((co * NCH + ci) * 3 + kh) * 3 + kw];
    float q  = rintf(wv / wmax * 7.0f) * (wmax / 7.0f);   // qmax = 2^(4-1)-1 = 7
    apack[tid] = (_Float16)q;
}

// ---------------------------------------------------------------------------
// Kernel 3: one workgroup = one output row (n,h), all 32 co, all 256 w.
// LDS holds 3 input rows, f16, layout [kh][x(+halo)][ci] -> ci innermost so a
// B fragment is one contiguous 32B read per lane. 8 waves x (1 co-tile x 4
// pixel-tiles) x 9 taps = 36 v_wmma_f32_16x16x32_f16 per wave.
// ---------------------------------------------------------------------------
__global__ __launch_bounds__(256) void conv_kernel(const float* __restrict__ x,
                                                   const _Float16* __restrict__ apack,
                                                   const float* __restrict__ bias,
                                                   float* __restrict__ out) {
    __shared__ _Float16 lds[3 * XW * NCH];   // 49,536 B

    const int n   = blockIdx.x >> 8;
    const int h   = blockIdx.x & 255;
    const int tid = threadIdx.x;

    // ---- zero the 1-column halos (xi = 0 and xi = 257, all kh, all ci) ----
    if (tid < 192) {
        int ci = tid & 31;
        int q  = tid >> 5;                 // 0..5
        int kh = q % 3;
        int xi = (q >= 3) ? (XW - 1) : 0;
        lds[(kh * XW + xi) * NCH + ci] = (_Float16)0.0f;
    }

    // ---- stage interior: 3 kh x 16 ci-pairs x 64 pixel-quads = 3072 iters,
    // exactly 12 per thread. Two float4 (b128) loads from adjacent ci planes,
    // v_cvt_pkrtz_f16_f32 packs an even/odd ci pair -> one b32 ds-store.
    // Consecutive lanes -> consecutive pixel quads -> 512B coalesced loads.
    for (int idx = tid; idx < 3 * 16 * 64; idx += 256) {
        int g    = idx & 63;               // pixel quad
        int rest = idx >> 6;
        int cp   = rest & 15;              // ci pair: channels 2cp, 2cp+1
        int kh   = rest >> 4;
        int r    = h + kh - 1;
        int w0   = g << 2;
        float4 va = make_float4(0.f, 0.f, 0.f, 0.f);
        float4 vb = va;
        if ((unsigned)r < (unsigned)HW) {
            va = *(const float4*)(x + (((n * NCH + 2 * cp    ) * HW) + r) * HW + w0);
            vb = *(const float4*)(x + (((n * NCH + 2 * cp + 1) * HW) + r) * HW + w0);
        }
        _Float16* dst = &lds[(kh * XW + w0 + 1) * NCH + 2 * cp];
        *(h2*)(dst          ) = __builtin_amdgcn_cvt_pkrtz(va.x, vb.x);
        *(h2*)(dst +     NCH) = __builtin_amdgcn_cvt_pkrtz(va.y, vb.y);
        *(h2*)(dst + 2 * NCH) = __builtin_amdgcn_cvt_pkrtz(va.z, vb.z);
        *(h2*)(dst + 3 * NCH) = __builtin_amdgcn_cvt_pkrtz(va.w, vb.w);
    }
    __syncthreads();

    const int wave   = tid >> 5;
    const int lane   = tid & 31;
    const int cotile = wave & 1;          // which 16 output channels
    const int ptbase = (wave >> 1) * 4;   // first of 4 pixel tiles
    const int cobase = cotile * 16;
    const int hi     = lane >> 4;         // lane half: 0 or 1
    const int lN     = lane & 15;         // N index within tile

    // ---- A fragments: 9 taps, pre-swizzled, 32B per lane each ----
    v16h a[9];
#pragma unroll
    for (int j = 0; j < 9; ++j)
        a[j] = *(const v16h*)(apack + (((cotile * 9 + j) * 32) + lane) * 16);

    // ---- per-lane bias for the 8 accumulator rows (C layout: M = v + 8*hi) --
    float bv[8];
#pragma unroll
    for (int v = 0; v < 8; ++v)
        bv[v] = bias[cobase + v + hi * 8];

#pragma unroll
    for (int pt = 0; pt < 4; ++pt) {
        const int px0 = (ptbase + pt) * 16;
        v8f acc = {};
#pragma unroll
        for (int j = 0; j < 9; ++j) {
            const int kh = j / 3, kw = j % 3;
            // B fragment: N = lN (pixel px0+lN), K = ci in [hi*16, hi*16+15],
            // input column = (px0+lN) + kw - 1 (+1 halo) = px0+lN+kw.
            const v16h b = *(const v16h*)(&lds[((kh * XW + px0 + lN + kw) * NCH) + hi * 16]);
            acc = __builtin_amdgcn_wmma_f32_16x16x32_f16(
                /*neg_a=*/false, a[j], /*neg_b=*/false, b,
                /*c_mod=*/(short)0, acc, /*reuse_a=*/false, /*reuse_b=*/false);
        }
        // ---- write D tile + bias: VGPR v, lane -> (co = cobase+v+8*hi, w = px0+lN)
#pragma unroll
        for (int v = 0; v < 8; ++v) {
            const int co = cobase + v + hi * 8;
            out[(((n * NCH + co) * HW) + h) * HW + px0 + lN] = acc[v] + bv[v];
        }
    }
}

// ---------------------------------------------------------------------------
extern "C" void kernel_launch(void* const* d_in, const int* in_sizes, int n_in,
                              void* d_out, int out_size, void* d_ws, size_t ws_size,
                              hipStream_t stream) {
    const float* x      = (const float*)d_in[0];   // (16,32,256,256)
    const float* weight = (const float*)d_in[1];   // (32,32,3,3)
    const float* bias   = (const float*)d_in[2];   // (32,)
    float* out = (float*)d_out;                    // (16,32,256,256)

    float*    wmaxp = (float*)d_ws;                          // 4 B
    _Float16* apack = (_Float16*)((char*)d_ws + 64);         // 18,432 B, 64B-aligned

    wmax_kernel<<<1, 256, 0, stream>>>(weight, wmaxp);
    pack_kernel<<<(NW + 255) / 256, 256, 0, stream>>>(weight, wmaxp, apack);
    conv_kernel<<<16 * HW, 256, 0, stream>>>(x, apack, bias, out);
}
